// CaptioningRNN_47854525612157
// MI455X (gfx1250) — compile-verified, matrix operand order checked by reading, not run on hardware
//
#include <hip/hip_runtime.h>
#include <hip/hip_bf16.h>
#include <math.h>

typedef __attribute__((ext_vector_type(16))) _Float16 v16h;
typedef __attribute__((ext_vector_type(8)))  float    v8f;
typedef int v4i __attribute__((vector_size(16)));

#define N_  64
#define T_  128
#define D_  512
#define H_  1024
#define FH_ 4096          // 4*H
#define KT_ 2560          // H (Wh) + H (Wattn) + D (Wx)
#define KP_ (KT_ / 2)     // 1280 k-pairs
#define KC_ 64            // K-chunk
#define NC_ (KT_ / KC_)   // 40 chunks
#define L_  16

#if defined(__has_builtin)
#if __has_builtin(__builtin_amdgcn_global_load_async_to_lds_b128)
#define HAS_ASYNC 1
#endif
#endif
#ifndef HAS_ASYNC
#define HAS_ASYNC 0
#endif

union U32H2 { unsigned int u; _Float16 h[2]; };
union FragLd { uint4 q[2]; v16h h; };   // forces 2x ds_load_b128 per fragment

__device__ __forceinline__ float fast_sigmoid(float x) {
  return 1.f / (1.f + __expf(-x));
}
__device__ __forceinline__ float fast_tanh(float x) {
  float xc = fminf(fmaxf(x, -15.f), 15.f);
  float e = __expf(-2.f * xc);
  return (1.f - e) / (1.f + e);
}

#if HAS_ASYNC
typedef __attribute__((address_space(1))) v4i* gv4i_p;
typedef __attribute__((address_space(3))) v4i* lv4i_p;
__device__ __forceinline__ void async_copy_b128(const void* g, void* l) {
  __builtin_amdgcn_global_load_async_to_lds_b128((gv4i_p)g, (lv4i_p)l, 0, 0);
}
template <int Nw>
__device__ __forceinline__ void wait_async() {
#if defined(__has_builtin) && __has_builtin(__builtin_amdgcn_s_wait_asynccnt)
  __builtin_amdgcn_s_wait_asynccnt(Nw);
#else
  asm volatile("s_wait_asynccnt %0" ::"i"(Nw) : "memory");
#endif
}
#endif

// ---------------- one-time conversion kernels ----------------

// Weights pre-packed in the WMMA B-fragment layout:
// Wpk[col * KP_ + kp] = pack_f16(W[2kp][col], W[2kp+1][col]),
// where W = [Wh ; Wattn ; Wx] (K = 2560 rows, 4096 cols).
__global__ void conv_wpk_kernel(const float* __restrict__ Wx,
                                const float* __restrict__ Wh,
                                const float* __restrict__ Wattn,
                                unsigned int* __restrict__ Wpk) {
  const int total = FH_ * KP_;
  for (int i = blockIdx.x * blockDim.x + threadIdx.x; i < total;
       i += gridDim.x * blockDim.x) {
    int col = i / KP_;
    int kp  = i - col * KP_;
    int r0  = 2 * kp;              // pair never straddles a region boundary
    float f0, f1;
    if (r0 < H_) {
      f0 = Wh[(size_t)r0 * FH_ + col];
      f1 = Wh[(size_t)(r0 + 1) * FH_ + col];
    } else if (r0 < 2 * H_) {
      f0 = Wattn[(size_t)(r0 - H_) * FH_ + col];
      f1 = Wattn[(size_t)(r0 - H_ + 1) * FH_ + col];
    } else {
      f0 = Wx[(size_t)(r0 - 2 * H_) * FH_ + col];
      f1 = Wx[(size_t)(r0 - 2 * H_ + 1) * FH_ + col];
    }
    U32H2 o;
    o.h[0] = (_Float16)f0;
    o.h[1] = (_Float16)f1;
    Wpk[i] = o.u;
  }
}

__global__ void conv_x_kernel(const float* __restrict__ x,
                              _Float16* __restrict__ xf) {
  const int total = N_ * T_ * D_;
  for (int i = blockIdx.x * blockDim.x + threadIdx.x; i < total;
       i += gridDim.x * blockDim.x) {
    xf[i] = (_Float16)x[i];
  }
}

__global__ void init_state_kernel(const float* __restrict__ A,
                                  float* __restrict__ h,
                                  float* __restrict__ c,
                                  _Float16* __restrict__ hcatA) {
  int i = blockIdx.x * blockDim.x + threadIdx.x;
  if (i >= N_ * H_) return;
  const float4* Ar = (const float4*)(A + (size_t)i * L_);
  float s = 0.f;
#pragma unroll
  for (int q = 0; q < 4; ++q) {
    float4 a4 = Ar[q];
    s += a4.x + a4.y + a4.z + a4.w;
  }
  float h0 = s * (1.f / 16.f);
  h[i] = h0;
  c[i] = h0;
  int n = i >> 10, j = i & (H_ - 1);
  hcatA[n * 2048 + j] = (_Float16)h0;
}

// ---------------- per-step attention ----------------
__global__ __launch_bounds__(256) void attn_kernel(
    const float* __restrict__ h,       // [64][1024]
    const float* __restrict__ A,       // [64][1024][16]
    _Float16* __restrict__ hcat) {     // write attn -> hcat[n][1024+j]
  __shared__ float red[256 * L_];
  const int n = blockIdx.x;
  const int tid = threadIdx.x;

  float p[L_];
#pragma unroll
  for (int l = 0; l < L_; ++l) p[l] = 0.f;

#pragma unroll
  for (int r = 0; r < 4; ++r) {
    int hh = tid + r * 256;
    float hv = h[n * H_ + hh];
    const float4* Ar = (const float4*)(A + ((size_t)n * H_ + hh) * L_);
#pragma unroll
    for (int q = 0; q < 4; ++q) {
      float4 a4 = Ar[q];
      p[4 * q + 0] += hv * a4.x;
      p[4 * q + 1] += hv * a4.y;
      p[4 * q + 2] += hv * a4.z;
      p[4 * q + 3] += hv * a4.w;
    }
  }
#pragma unroll
  for (int l = 0; l < L_; ++l) red[tid * L_ + l] = p[l];
  __syncthreads();
  for (int s = 128; s > 0; s >>= 1) {
    if (tid < s) {
#pragma unroll
      for (int l = 0; l < L_; ++l)
        red[tid * L_ + l] += red[(tid + s) * L_ + l];
    }
    __syncthreads();
  }

  const float scale = 0.03125f;  // 1/sqrt(1024)
  float sc[L_], m = -3.0e38f;
#pragma unroll
  for (int l = 0; l < L_; ++l) {
    sc[l] = red[l] * scale;
    m = fmaxf(m, sc[l]);
  }
  float sum = 0.f;
#pragma unroll
  for (int l = 0; l < L_; ++l) {
    sc[l] = __expf(sc[l] - m);
    sum += sc[l];
  }
  float inv = 1.f / sum;
#pragma unroll
  for (int l = 0; l < L_; ++l) sc[l] *= inv;

#pragma unroll
  for (int r = 0; r < 4; ++r) {
    int hh = tid + r * 256;
    const float4* Ar = (const float4*)(A + ((size_t)n * H_ + hh) * L_);
    float at = 0.f;
#pragma unroll
    for (int q = 0; q < 4; ++q) {
      float4 a4 = Ar[q];
      at += a4.x * sc[4 * q + 0] + a4.y * sc[4 * q + 1] +
            a4.z * sc[4 * q + 2] + a4.w * sc[4 * q + 3];
    }
    hcat[n * 2048 + H_ + hh] = (_Float16)at;
  }
}

// ---------------- per-step fused GEMM + LSTM gates ----------------
// 32 blocks x 8 waves; block owns 32 H-columns across all 4 gate regions
// (64x128 tile); wave = (gate, mhalf) computes 32x32 = 2x2 WMMA fragments.
// Async path: triple-buffered LDS ring fed by global_load_async_to_lds_b128
// (no staging VGPRs), one barrier per K-chunk of 64.
__attribute__((amdgpu_waves_per_eu(2, 8)))
__global__ __launch_bounds__(256) void step_kernel(
    const _Float16* __restrict__ hcatIn,   // [64][2048]  (h | attn) f16
    _Float16* __restrict__ hcatOut,        // write next h -> [n][0..1023]
    const _Float16* __restrict__ xf,       // [N][T][D] f16
    const unsigned int* __restrict__ Wpk,  // [4096 cols][1280 k-pairs]
    const float* __restrict__ bias,        // [4096]
    float* __restrict__ c,                 // [64][1024]
    float* __restrict__ h,                 // [64][1024]
    float* __restrict__ out,               // [N][T][H]
    int t) {
#if HAS_ASYNC
  // 3 x (A 8KB + B 16KB) = 72KB ring; epilogue reuses first 32KB as aTile.
  __shared__ __align__(16) unsigned char smem[73728];
#else
  // 2 x (A 8KB + B 16KB) = 48KB; epilogue reuses first 32KB as aTile.
  __shared__ __align__(16) unsigned char smem[49152];
#endif
  float* aTile = (float*)smem;

  const int tid   = threadIdx.x;
  const int lane  = tid & 31;
  const int wave  = tid >> 5;
  const int gate  = wave & 3;
  const int mhalf = wave >> 2;
  const int j0    = blockIdx.x * 32;

  const int khalf = lane >> 4;
  const int lrow  = lane & 15;

  const v8f zero = {0.f, 0.f, 0.f, 0.f, 0.f, 0.f, 0.f, 0.f};
  v8f acc[2][2];
  acc[0][0] = zero; acc[0][1] = zero; acc[1][0] = zero; acc[1][1] = zero;

  // staging assignments (pure copies)
  const int an  = tid >> 2;            // A row 0..63
  const int ak  = (tid & 3) * 16;      // A k-offset within chunk (16 f16)
  const int colL  = tid & 127;         // B local col 0..127
  const int bhalf = tid >> 7;          // which 16 k-pairs of the chunk
  const int gcol  = (colL >> 5) * H_ + j0 + (colL & 31);  // global col
  const unsigned int* wcol = Wpk + (size_t)gcol * KP_ + bhalf * 16;

#if HAS_ASYNC
  // ---------------- async-DMA triple-buffer pipeline ----------------
  auto issue_chunk = [&](int ic) {
    const int kc = ic * KC_;
    const int b3 = ic % 3;
    _Float16*     bA  = (_Float16*)(smem + b3 * 8192);
    unsigned int* bBu = (unsigned int*)(smem + 24576 + b3 * 16384);
    int k = kc + ak;
    const _Float16* src =
        (k < 2048) ? (hcatIn + an * 2048 + k)
                   : (xf + ((size_t)(an * T_ + t)) * D_ + (k - 2048));
    _Float16* dA = bA + an * 64 + ak;
    async_copy_b128(src, dA);
    async_copy_b128(src + 8, dA + 8);
    const unsigned int* w = wcol + (kc >> 1);
    unsigned int* dB = bBu + colL * 32 + bhalf * 16;
#pragma unroll
    for (int r = 0; r < 4; ++r) async_copy_b128(w + 4 * r, dB + 4 * r);
  };

  issue_chunk(0);
  for (int ic = 0; ic < NC_; ++ic) {
    const int b3 = ic % 3;
    const unsigned int* bAu = (const unsigned int*)(smem + b3 * 8192);
    const unsigned int* bBu =
        (const unsigned int*)(smem + 24576 + b3 * 16384);

    if (ic + 1 < NC_) {
      issue_chunk(ic + 1);      // 6 async b128 copies, in flight over compute
      wait_async<6>();          // in-order: chunk ic's 6 copies have landed
    } else {
      wait_async<0>();
    }
    if (ic + 2 < NC_) __builtin_prefetch(wcol + (((ic + 2) * KC_) >> 1), 0, 1);
    __syncthreads();            // chunk ic visible to all waves

#pragma unroll
    for (int s = 0; s < 2; ++s) {
      FragLd af[2], bf[2];
#pragma unroll
      for (int i = 0; i < 2; ++i) {
        const unsigned int* pa =
            bAu + (mhalf * 32 + i * 16 + lrow) * 32 + s * 16 + khalf * 4;
        af[i].q[0] = *(const uint4*)pa;
        af[i].q[1] = *(const uint4*)(pa + 8);
      }
#pragma unroll
      for (int j = 0; j < 2; ++j) {
        const unsigned int* pb =
            bBu + (gate * 32 + j * 16 + lrow) * 32 + s * 16 + khalf * 4;
        bf[j].q[0] = *(const uint4*)pb;
        bf[j].q[1] = *(const uint4*)(pb + 8);
      }
#pragma unroll
      for (int i = 0; i < 2; ++i)
#pragma unroll
        for (int j = 0; j < 2; ++j)
          acc[i][j] = __builtin_amdgcn_wmma_f32_16x16x32_f16(
              false, af[i].h, false, bf[j].h, (short)0, acc[i][j], false,
              false);
    }
  }
#else
  // ---------------- register-staged double-buffer fallback ----------------
  uint4 ra[2], rb[4];
  {
    const _Float16* src = hcatIn + an * 2048 + ak;
    ra[0] = *(const uint4*)src;
    ra[1] = *(const uint4*)(src + 8);
#pragma unroll
    for (int r = 0; r < 4; ++r) rb[r] = *(const uint4*)(wcol + 4 * r);
  }
  for (int ic = 0; ic < NC_; ++ic) {
    const int buf = ic & 1;
    _Float16*     bA  = (_Float16*)(smem + buf * 8192);
    const unsigned int* bAu = (const unsigned int*)bA;
    unsigned int* bBu = (unsigned int*)(smem + 16384 + buf * 16384);

    *(uint4*)(bA + an * 64 + ak)     = ra[0];
    *(uint4*)(bA + an * 64 + ak + 8) = ra[1];
#pragma unroll
    for (int r = 0; r < 4; ++r)
      *(uint4*)(bBu + colL * 32 + bhalf * 16 + 4 * r) = rb[r];

    if (ic + 1 < NC_) {
      const int kc = (ic + 1) * KC_;
      int k = kc + ak;
      const _Float16* src =
          (k < 2048) ? (hcatIn + an * 2048 + k)
                     : (xf + ((size_t)(an * T_ + t)) * D_ + (k - 2048));
      ra[0] = *(const uint4*)src;
      ra[1] = *(const uint4*)(src + 8);
      const unsigned int* w = wcol + (kc >> 1);
#pragma unroll
      for (int r = 0; r < 4; ++r) rb[r] = *(const uint4*)(w + 4 * r);
    }
    if (ic + 2 < NC_) __builtin_prefetch(wcol + (((ic + 2) * KC_) >> 1), 0, 1);
    __syncthreads();

#pragma unroll
    for (int s = 0; s < 2; ++s) {
      FragLd af[2], bf[2];
#pragma unroll
      for (int i = 0; i < 2; ++i) {
        const unsigned int* pa =
            bAu + (mhalf * 32 + i * 16 + lrow) * 32 + s * 16 + khalf * 4;
        af[i].q[0] = *(const uint4*)pa;
        af[i].q[1] = *(const uint4*)(pa + 8);
      }
#pragma unroll
      for (int j = 0; j < 2; ++j) {
        const unsigned int* pb =
            bBu + (gate * 32 + j * 16 + lrow) * 32 + s * 16 + khalf * 4;
        bf[j].q[0] = *(const uint4*)pb;
        bf[j].q[1] = *(const uint4*)(pb + 8);
      }
#pragma unroll
      for (int i = 0; i < 2; ++i)
#pragma unroll
        for (int j = 0; j < 2; ++j)
          acc[i][j] = __builtin_amdgcn_wmma_f32_16x16x32_f16(
              false, af[i].h, false, bf[j].h, (short)0, acc[i][j], false,
              false);
    }
  }
#endif

  __syncthreads();  // all fragment ds-loads done before aTile overwrite

  // ---- spill accumulators to LDS ----
#pragma unroll
  for (int i = 0; i < 2; ++i) {
#pragma unroll
    for (int j = 0; j < 2; ++j) {
      int colW  = gate * 32 + j * 16 + lrow;
      int rbase = mhalf * 32 + i * 16 + khalf * 8;
#pragma unroll
      for (int e = 0; e < 8; ++e)
        aTile[(rbase + e) * 128 + colW] = acc[i][j][e];
    }
  }
  __syncthreads();

  // ---- fused gate math + state update (branchless activations) ----
  {
    int n   = tid >> 2;
    int cg2 = (tid & 3) * 8;
#pragma unroll
    for (int q = 0; q < 8; ++q) {
      int col = cg2 + q;
      int j   = j0 + col;
      float av_i = aTile[n * 128 +  0 + col] + bias[j];
      float av_f = aTile[n * 128 + 32 + col] + bias[H_ + j];
      float av_o = aTile[n * 128 + 64 + col] + bias[2 * H_ + j];
      float av_g = aTile[n * 128 + 96 + col] + bias[3 * H_ + j];
      float ig = fast_sigmoid(av_i);
      float fg = fast_sigmoid(av_f);
      float og = fast_sigmoid(av_o);
      float gg = fast_tanh(av_g);
      float cn = fg * c[n * H_ + j] + ig * gg;
      float hn = og * fast_tanh(cn);
      c[n * H_ + j] = cn;
      h[n * H_ + j] = hn;
      hcatOut[n * 2048 + j] = (_Float16)hn;
      out[((size_t)n * T_ + t) * H_ + j] = hn;
    }
  }
}

// ---------------- host-side launch ----------------

extern "C" void kernel_launch(void* const* d_in, const int* in_sizes, int n_in,
                              void* d_out, int out_size, void* d_ws,
                              size_t ws_size, hipStream_t stream) {
  (void)in_sizes; (void)n_in; (void)out_size; (void)ws_size;
  const float* x     = (const float*)d_in[0];
  const float* A     = (const float*)d_in[1];
  const float* Wx    = (const float*)d_in[2];
  const float* Wh    = (const float*)d_in[3];
  const float* Wattn = (const float*)d_in[4];
  const float* b     = (const float*)d_in[5];
  float* out = (float*)d_out;

  char* ws = (char*)d_ws;
  size_t off = 0;
  auto wsAlloc = [&](size_t bytes) -> void* {
    void* p = ws + off;
    off = (off + bytes + 255) & ~(size_t)255;
    return p;
  };
  unsigned int* Wpk =
      (unsigned int*)wsAlloc((size_t)FH_ * KP_ * sizeof(unsigned int));
  _Float16* xf    = (_Float16*)wsAlloc((size_t)N_ * T_ * D_ * sizeof(_Float16));
  _Float16* hcatA = (_Float16*)wsAlloc((size_t)N_ * 2048 * sizeof(_Float16));
  _Float16* hcatB = (_Float16*)wsAlloc((size_t)N_ * 2048 * sizeof(_Float16));
  float* hbuf = (float*)wsAlloc((size_t)N_ * H_ * sizeof(float));
  float* cbuf = (float*)wsAlloc((size_t)N_ * H_ * sizeof(float));

  conv_wpk_kernel<<<4096, 256, 0, stream>>>(Wx, Wh, Wattn, Wpk);
  conv_x_kernel<<<4096, 256, 0, stream>>>(x, xf);
  init_state_kernel<<<(N_ * H_ + 255) / 256, 256, 0, stream>>>(A, hbuf, cbuf,
                                                               hcatA);

  for (int t = 0; t < T_; ++t) {
    _Float16* hr = (t & 1) ? hcatB : hcatA;  // read (h_{t-1} | attn_t)
    _Float16* hw = (t & 1) ? hcatA : hcatB;  // write h_t (ping-pong)
    attn_kernel<<<N_, 256, 0, stream>>>(hbuf, A, hr);
    step_kernel<<<H_ / 32, 256, 0, stream>>>(hr, hw, xf, Wpk, b, cbuf, hbuf,
                                             out, t);
  }
}
